// VideoSwinBlock_20847771255238
// MI455X (gfx1250) — matrix operand order repaired and checked by reading, not compiled, not run on hardware
//
#include <hip/hip_runtime.h>
#include <hip/hip_bf16.h>

typedef _Float16 v8h  __attribute__((ext_vector_type(8)));
typedef _Float16 v16h __attribute__((ext_vector_type(16)));
typedef float    v8f  __attribute__((ext_vector_type(8)));

#define SEQ    49
#define CH     384
#define NHEAD  12
#define HDIM   32
#define HID    1536
#define QK_LD  768
#define NWAVE  8

union V16U { v16h v; v8h h[2]; };

// A fragment: 16x32 f16, row M = lane&15, K-half selected by lane&16.
__device__ __forceinline__ v16h load_a_frag(const _Float16* m, int ld, int row0, int k0, int lane) {
    const int M  = lane & 15;
    const int kb = k0 + ((lane & 16) ? 16 : 0);
    const _Float16* p = m + (row0 + M) * ld + kb;
    V16U u;
    u.h[0] = *(const v8h*)(p);
    u.h[1] = *(const v8h*)(p + 8);
    return u.v;
}

// B fragment: 32x16 f16, col N = lane&15, lanes 0-15 hold K=k0..k0+15,
// lanes 16-31 hold K=k0+16..k0+31 -> 32 contiguous bytes of row (n0+N).
__device__ __forceinline__ v16h load_b_frag(const _Float16* m, int ld, int n0, int k0, int lane) {
    const int N  = lane & 15;
    const int kb = k0 + ((lane & 16) ? 16 : 0);
    const _Float16* p = m + (n0 + N) * ld + kb;
    V16U u;
    u.h[0] = *(const v8h*)(p);
    u.h[1] = *(const v8h*)(p + 8);
    return u.v;
}

__device__ __forceinline__ v8f wmma16(v16h a, v16h b, v8f c) {
    return __builtin_amdgcn_wmma_f32_16x16x32_f16(false, a, false, b, (short)0, c, false, false);
}

__device__ __forceinline__ void layernorm_rows(const float* __restrict__ src,
                                               _Float16* __restrict__ dst,
                                               const float* __restrict__ g,
                                               const float* __restrict__ b,
                                               int lane, int wid) {
    for (int row = wid; row < SEQ; row += NWAVE) {
        const float* r = src + row * CH;
        float x[12];
        float s = 0.f;
#pragma unroll
        for (int j = 0; j < 12; ++j) { x[j] = r[j * 32 + lane]; s += x[j]; }
#pragma unroll
        for (int m = 16; m >= 1; m >>= 1) s += __shfl_xor(s, m, 32);
        const float mean = s * (1.0f / CH);
        float v = 0.f;
#pragma unroll
        for (int j = 0; j < 12; ++j) { const float d = x[j] - mean; v += d * d; }
#pragma unroll
        for (int m = 16; m >= 1; m >>= 1) v += __shfl_xor(v, m, 32);
        const float rstd = rsqrtf(v * (1.0f / CH) + 1e-5f);
        _Float16* o = dst + row * CH;
#pragma unroll
        for (int j = 0; j < 12; ++j) {
            const int c = j * 32 + lane;
            o[c] = (_Float16)((x[j] - mean) * rstd * g[c] + b[c]);
        }
    }
}

__global__ void cvt_f32_to_f16(const float* __restrict__ src, _Float16* __restrict__ dst, int n) {
    const int i = blockIdx.x * blockDim.x + threadIdx.x;
    if (i < n) dst[i] = (_Float16)src[i];
}

__launch_bounds__(256, 1)
__global__ void swin_block_kernel(const float* __restrict__ tokens,
                                  const _Float16* __restrict__ w16,
                                  const float* __restrict__ qkv_b,
                                  const float* __restrict__ out_b,
                                  const float* __restrict__ ln1_g, const float* __restrict__ ln1_b,
                                  const float* __restrict__ b1,
                                  const float* __restrict__ b2,
                                  const float* __restrict__ ln2_g, const float* __restrict__ ln2_b,
                                  float* __restrict__ outp) {
    __shared__ __align__(16) float    resid[SEQ * CH];        // 75,264 B  residual (f32)
    __shared__ __align__(16) _Float16 a16[64 * CH];           // 49,152 B  LN out; reused as attn O
    __shared__ __align__(16) _Float16 qkv[64 * 1152];         // 147,456 B q,k (64x768) | vT (384x64);
                                                              //            later: f32 mlpacc + f16 h1
    __shared__ __align__(16) _Float16 pscr[NWAVE * 16 * 64];  // 16,384 B  per-wave softmax P scratch

    const int tid  = threadIdx.x;
    const int lane = tid & 31;
    const int wid  = tid >> 5;

    // window decode: grid = (b*t) * 8 * 8 = 1024
    const int bt  = blockIdx.x >> 6;
    const int win = blockIdx.x & 63;
    const int wh  = win >> 3, ww = win & 7;
    const size_t gbase = ((size_t)bt * 3136 + (size_t)(wh * 7) * 56 + (size_t)(ww * 7)) * CH;
    const float* gin  = tokens + gbase;
    float*       gout = outp   + gbase;

    const _Float16* wqkv = w16;
    const _Float16* wout = w16 + 1152 * 384;
    const _Float16* w1p  = w16 + 1152 * 384 + 384 * 384;
    const _Float16* w2p  = w16 + 1152 * 384 + 384 * 384 + 1536 * 384;

    const int cb   = lane & 15;
    const int rsel = (lane & 16) ? 8 : 0;

    // GEMM wave mapping: 2 M-tiles x 2 N-tiles per wave, 4 accumulator chains.
    const int mg  = wid & 1;
    const int mtA = mg * 2, mtB = mg * 2 + 1;
    const int nq  = wid >> 1;  // n-pair stride-4 start

    // ---- Phase 0: async DMA of the window (7x7 tokens x 384ch) into LDS ----
    for (int i = tid; i < SEQ * (CH / 4); i += 256) {
        const int m  = i / 96;
        const int c4 = (i % 96) * 4;
        const float*   gsrc    = gin + (size_t)((m / 7) * 56 + (m % 7)) * CH + c4;
        const unsigned lds_off = (unsigned)(uintptr_t)(resid + m * CH + c4);
        asm volatile("global_load_async_to_lds_b128 %0, %1, off"
                     :: "v"(lds_off), "v"(gsrc) : "memory");
    }
    asm volatile("s_wait_asynccnt 0" ::: "memory");
    __syncthreads();

    // ---- Phase 1: LayerNorm1 -> a16 ; zero pad rows 49..63 (keep K/V finite) ----
    layernorm_rows(resid, a16, ln1_g, ln1_b, lane, wid);
    for (int i = tid; i < (64 - SEQ) * CH; i += 256) a16[SEQ * CH + i] = (_Float16)0.f;
    __syncthreads();

    // ---- Phase 2: QKV GEMM (64x384 @ 384x1152) -> q,k row-major ; V transposed ----
    {
        v16h afr[12];
#pragma unroll
        for (int ks = 0; ks < 12; ++ks) afr[ks] = load_a_frag(a16, CH, mtA * 16, ks * 32, lane);

        auto store_qkv = [&](const v8f& acc, int mt, int nt) {
            const int col  = nt * 16 + cb;
            const float bv = qkv_b[col];
            const int rb   = mt * 16 + rsel;
            if (nt < 48) {  // q,k: row-major, ld = 768
#pragma unroll
                for (int r = 0; r < 8; ++r)
                    qkv[(rb + r) * QK_LD + col] = (_Float16)(acc[r] + bv);
            } else {        // v: store transposed -> vT[channel][token], ld = 64
                _Float16* vT = qkv + 64 * QK_LD;
                const int vrow = col - 768;
#pragma unroll
                for (int r = 0; r < 8; ++r)
                    vT[vrow * 64 + (rb + r)] = (_Float16)(acc[r] + bv);
            }
        };

        for (int np = nq; np < 36; np += 4) {
            const int nt0 = np * 2, nt1 = nt0 + 1;
            __builtin_prefetch(wqkv + (size_t)((np + 4) * 32 + cb) * CH, 0, 3);
            v8f a00 = {}, a01 = {}, a10 = {}, a11 = {};
#pragma unroll
            for (int ks = 0; ks < 12; ++ks) {
                const v16h b0 = load_b_frag(wqkv, CH, nt0 * 16, ks * 32, lane);
                const v16h b1 = load_b_frag(wqkv, CH, nt1 * 16, ks * 32, lane);
                const v16h a1 = load_a_frag(a16, CH, mtB * 16, ks * 32, lane);
                a00 = wmma16(afr[ks], b0, a00);
                a01 = wmma16(afr[ks], b1, a01);
                a10 = wmma16(a1, b0, a10);
                a11 = wmma16(a1, b1, a11);
            }
            store_qkv(a00, mtA, nt0); store_qkv(a01, mtA, nt1);
            store_qkv(a10, mtB, nt0); store_qkv(a11, mtB, nt1);
        }
    }
    __syncthreads();

    // ---- Phase 3: attention; 48 (head, m-tile) jobs -> 6 per wave (balanced) ----
    {
        _Float16* o16 = a16;  // reuse (LN1 activations are dead)
        const _Float16* vT = qkv + 64 * QK_LD;
        _Float16* pp = pscr + wid * (16 * 64);
        for (int job = wid; job < NHEAD * 4; job += NWAVE) {
            const int head = job >> 2;
            const int mt   = job & 3;
            const int ho   = head * HDIM;
            // scores = q @ k^T : K=32, one WMMA per 16x16 tile
            const v16h qa = load_a_frag(qkv, QK_LD, mt * 16, ho, lane);
            v8f sc[4];
#pragma unroll
            for (int nt = 0; nt < 4; ++nt) {
                v8f z = {};
                sc[nt] = wmma16(qa, load_b_frag(qkv + CH + ho, QK_LD, nt * 16, 0, lane), z);
            }
            // scale + mask (select, never multiply: avoid NaN*0 from pad K rows)
#pragma unroll
            for (int nt = 0; nt < 4; ++nt) {
                const int col = nt * 16 + cb;
#pragma unroll
                for (int r = 0; r < 8; ++r)
                    sc[nt][r] = (col < SEQ) ? sc[nt][r] * 0.17677669529663687f : -1e30f;
            }
            // row softmax: reduce across the 16 lanes of each half (xor 1/2/4/8)
#pragma unroll
            for (int r = 0; r < 8; ++r) {
                float mx = fmaxf(fmaxf(sc[0][r], sc[1][r]), fmaxf(sc[2][r], sc[3][r]));
#pragma unroll
                for (int m = 8; m >= 1; m >>= 1) mx = fmaxf(mx, __shfl_xor(mx, m, 32));
                float sum = 0.f;
#pragma unroll
                for (int nt = 0; nt < 4; ++nt) {
                    const float e = __expf(sc[nt][r] - mx);
                    sc[nt][r] = e;
                    sum += e;
                }
#pragma unroll
                for (int m = 8; m >= 1; m >>= 1) sum += __shfl_xor(sum, m, 32);
                const float inv = 1.0f / sum;
#pragma unroll
                for (int nt = 0; nt < 4; ++nt) sc[nt][r] *= inv;
            }
            // P (16x64 f16) -> wave-private scratch
#pragma unroll
            for (int nt = 0; nt < 4; ++nt)
#pragma unroll
                for (int r = 0; r < 8; ++r)
                    pp[(r + rsel) * 64 + nt * 16 + cb] = (_Float16)sc[nt][r];
            // O = P @ V : vT rows are channels -> contiguous B fragments
#pragma unroll
            for (int nt2 = 0; nt2 < 2; ++nt2) {
                v8f oacc = {};
#pragma unroll
                for (int ks = 0; ks < 2; ++ks)
                    oacc = wmma16(load_a_frag(pp, 64, 0, ks * 32, lane),
                                  load_b_frag(vT, 64, ho + nt2 * 16, ks * 32, lane), oacc);
#pragma unroll
                for (int r = 0; r < 8; ++r) {
                    const int row = mt * 16 + r + rsel;
                    o16[row * CH + ho + nt2 * 16 + cb] = (_Float16)oacc[r];
                }
            }
        }
    }
    __syncthreads();

    // ---- Phase 4: out projection + residual into resid (f32) ----
    {
        const _Float16* o16 = a16;
        v16h afr[12];
#pragma unroll
        for (int ks = 0; ks < 12; ++ks) afr[ks] = load_a_frag(o16, CH, mtA * 16, ks * 32, lane);

        auto store_out = [&](const v8f& acc, int mt, int nt) {
            const int col  = nt * 16 + cb;
            const float bv = out_b[col];
            const int rb   = mt * 16 + rsel;
#pragma unroll
            for (int r = 0; r < 8; ++r) {
                const int row = rb + r;
                if (row < SEQ) resid[row * CH + col] += acc[r] + bv;
            }
        };

        for (int np = nq; np < 12; np += 4) {
            const int nt0 = np * 2, nt1 = nt0 + 1;
            __builtin_prefetch(wout + (size_t)((np + 4) * 32 + cb) * CH, 0, 3);
            v8f a00 = {}, a01 = {}, a10 = {}, a11 = {};
#pragma unroll
            for (int ks = 0; ks < 12; ++ks) {
                const v16h b0 = load_b_frag(wout, CH, nt0 * 16, ks * 32, lane);
                const v16h b1 = load_b_frag(wout, CH, nt1 * 16, ks * 32, lane);
                const v16h a1 = load_a_frag(o16, CH, mtB * 16, ks * 32, lane);
                a00 = wmma16(afr[ks], b0, a00);
                a01 = wmma16(afr[ks], b1, a01);
                a10 = wmma16(a1, b0, a10);
                a11 = wmma16(a1, b1, a11);
            }
            store_out(a00, mtA, nt0); store_out(a01, mtA, nt1);
            store_out(a10, mtB, nt0); store_out(a11, mtB, nt1);
        }
    }
    __syncthreads();

    // ---- Phase 5: LayerNorm2 -> a16 ----
    layernorm_rows(resid, a16, ln2_g, ln2_b, lane, wid);
    __syncthreads();

    // ---- Phase 6: MLP in 4 chunks of HID=384 ; fc2 accumulates into f32 LDS;
    //      last chunk fuses residual + bias and stores straight to global ----
    {
        float*    mlpacc = (float*)(void*)qkv;   // 64x384 f32 (q/k/vT dead)
        _Float16* h1     = qkv + 64 * CH * 2;    // 64x384 f16 hidden chunk
        for (int chk = 0; chk < 4; ++chk) {
            // fc1 chunk: gelu(a16 @ w1^T[:, chk*384 : +384] + b1)
            {
                v16h afr[12];
#pragma unroll
                for (int ks = 0; ks < 12; ++ks) afr[ks] = load_a_frag(a16, CH, mtA * 16, ks * 32, lane);

                auto store_h1 = [&](const v8f& acc, int mt, int nt) {
                    const int col  = nt * 16 + cb;
                    const float bv = b1[chk * CH + col];
                    const int rb   = mt * 16 + rsel;
#pragma unroll
                    for (int r = 0; r < 8; ++r) {
                        float x = acc[r] + bv;
                        x = 0.5f * x * (1.0f + erff(x * 0.70710678118f));   // exact gelu
                        h1[(rb + r) * CH + col] = (_Float16)x;
                    }
                };

                for (int np = nq; np < 12; np += 4) {
                    const int nt0 = np * 2, nt1 = nt0 + 1;
                    __builtin_prefetch(w1p + (size_t)(chk * CH + (np + 4) * 32 + cb) * CH, 0, 3);
                    v8f a00 = {}, a01 = {}, a10 = {}, a11 = {};
#pragma unroll
                    for (int ks = 0; ks < 12; ++ks) {
                        const v16h b0 = load_b_frag(w1p, CH, chk * CH + nt0 * 16, ks * 32, lane);
                        const v16h b1f = load_b_frag(w1p, CH, chk * CH + nt1 * 16, ks * 32, lane);
                        const v16h a1 = load_a_frag(a16, CH, mtB * 16, ks * 32, lane);
                        a00 = wmma16(afr[ks], b0, a00);
                        a01 = wmma16(afr[ks], b1f, a01);
                        a10 = wmma16(a1, b0, a10);
                        a11 = wmma16(a1, b1f, a11);
                    }
                    store_h1(a00, mtA, nt0); store_h1(a01, mtA, nt1);
                    store_h1(a10, mtB, nt0); store_h1(a11, mtB, nt1);
                }
            }
            __syncthreads();
            // fc2 partial: mlpacc += h1 @ w2^T[chk*384 : +384, :]
            // chk==3: out = resid + mlpacc + acc + b2 -> global (window scatter)
            {
                v16h afr[12];
#pragma unroll
                for (int ks = 0; ks < 12; ++ks) afr[ks] = load_a_frag(h1, CH, mtA * 16, ks * 32, lane);

                auto store_acc = [&](const v8f& acc, int mt, int nt) {
                    const int col = nt * 16 + cb;
                    const int rb  = mt * 16 + rsel;
                    if (chk < 3) {
#pragma unroll
                        for (int r = 0; r < 8; ++r) {
                            float* d = mlpacc + (rb + r) * CH + col;
                            if (chk == 0) *d = acc[r]; else *d += acc[r];
                        }
                    } else {
                        const float bv = b2[col];
#pragma unroll
                        for (int r = 0; r < 8; ++r) {
                            const int row = rb + r;
                            if (row < SEQ) {
                                const float v = resid[row * CH + col] +
                                                mlpacc[(rb + r) * CH + col] + acc[r] + bv;
                                gout[(size_t)((row / 7) * 56 + (row % 7)) * CH + col] = v;
                            }
                        }
                    }
                };

                for (int np = nq; np < 12; np += 4) {
                    const int nt0 = np * 2, nt1 = nt0 + 1;
                    __builtin_prefetch(w2p + (size_t)(chk * CH) + (size_t)((np + 4) * 32 + cb) * HID, 0, 3);
                    v8f a00 = {}, a01 = {}, a10 = {}, a11 = {};
#pragma unroll
                    for (int ks = 0; ks < 12; ++ks) {
                        const v16h b0 = load_b_frag(w2p + chk * CH, HID, nt0 * 16, ks * 32, lane);
                        const v16h b1f = load_b_frag(w2p + chk * CH, HID, nt1 * 16, ks * 32, lane);
                        const v16h a1 = load_a_frag(h1, CH, mtB * 16, ks * 32, lane);
                        a00 = wmma16(afr[ks], b0, a00);
                        a01 = wmma16(afr[ks], b1f, a01);
                        a10 = wmma16(a1, b0, a10);
                        a11 = wmma16(a1, b1f, a11);
                    }
                    store_acc(a00, mtA, nt0); store_acc(a01, mtA, nt1);
                    store_acc(a10, mtB, nt0); store_acc(a11, mtB, nt1);
                }
            }
            __syncthreads();
        }
    }
}

extern "C" void kernel_launch(void* const* d_in, const int* in_sizes, int n_in,
                              void* d_out, int out_size, void* d_ws, size_t ws_size,
                              hipStream_t stream) {
    (void)in_sizes; (void)n_in; (void)out_size; (void)ws_size;
    const float* tokens    = (const float*)d_in[0];
    const float* in_proj_w = (const float*)d_in[1];
    const float* in_proj_b = (const float*)d_in[2];
    const float* out_w     = (const float*)d_in[3];
    const float* out_b     = (const float*)d_in[4];
    const float* ln1_g     = (const float*)d_in[5];
    const float* ln1_b     = (const float*)d_in[6];
    const float* w1        = (const float*)d_in[7];
    const float* b1        = (const float*)d_in[8];
    const float* w2        = (const float*)d_in[9];
    const float* b2        = (const float*)d_in[10];
    const float* ln2_g     = (const float*)d_in[11];
    const float* ln2_b     = (const float*)d_in[12];

    _Float16* w16 = (_Float16*)d_ws;
    const int nqkv = 1152 * 384, nout = 384 * 384, nw1 = 1536 * 384, nw2 = 384 * 1536;
    cvt_f32_to_f16<<<(nqkv + 255) / 256, 256, 0, stream>>>(in_proj_w, w16, nqkv);
    cvt_f32_to_f16<<<(nout + 255) / 256, 256, 0, stream>>>(out_w, w16 + nqkv, nout);
    cvt_f32_to_f16<<<(nw1 + 255) / 256, 256, 0, stream>>>(w1, w16 + nqkv + nout, nw1);
    cvt_f32_to_f16<<<(nw2 + 255) / 256, 256, 0, stream>>>(w2, w16 + nqkv + nout + nw1, nw2);

    swin_block_kernel<<<1024, 256, 0, stream>>>(tokens, w16, in_proj_b, out_b,
                                                ln1_g, ln1_b, b1, b2, ln2_g, ln2_b,
                                                (float*)d_out);
}
